// StackedLinear_25228637896975
// MI455X (gfx1250) — compile-verified, matrix operand order checked by reading, not run on hardware
//
#include <hip/hip_runtime.h>

#define IN_FEATURES 1024
#define OUT_FEATURES 1024
#define COUNT 16
#define BATCH 8192

typedef __attribute__((ext_vector_type(16))) __bf16          v16bf;
typedef __attribute__((ext_vector_type(16))) unsigned short  v16us;
typedef __attribute__((ext_vector_type(8)))  unsigned short  v8us;
typedef __attribute__((ext_vector_type(4)))  unsigned short  v4us;
typedef __attribute__((ext_vector_type(8)))  float           v8f;

// f32 -> bf16 round-to-nearest-even (bit trick; inputs are finite gaussians)
__device__ __forceinline__ unsigned short f32_to_bf16(float f) {
    unsigned int u = __builtin_bit_cast(unsigned int, f);
    u += 0x7fffu + ((u >> 16) & 1u);
    return (unsigned short)(u >> 16);
}
__device__ __forceinline__ float bf16_to_f32(unsigned short h) {
    unsigned int u = ((unsigned int)h) << 16;
    return __builtin_bit_cast(float, u);
}

__device__ __forceinline__ v16bf mk_frag(v8us a, v8us b) {
    v16us v = __builtin_shufflevector(a, b, 0, 1, 2, 3, 4, 5, 6, 7,
                                            8, 9, 10, 11, 12, 13, 14, 15);
    return __builtin_bit_cast(v16bf, v);
}

__global__ void zero_counts_kernel(int* __restrict__ counts) {
    if (threadIdx.x < COUNT) counts[threadIdx.x] = 0;
}

__global__ void bin_rows_kernel(const int* __restrict__ ls_indices,
                                int* __restrict__ counts,
                                int* __restrict__ perm) {
    int i = blockIdx.x * blockDim.x + threadIdx.x;
    if (i < BATCH) {
        int e = ls_indices[i] & (COUNT - 1);
        int pos = atomicAdd(&counts[e], 1);
        perm[e * BATCH + pos] = i;
    }
}

// One-shot bf16x2 split of an f32 array: src = hi + lo (each bf16), 4 elems/thread.
__global__ __launch_bounds__(256)
void split_bf16_kernel(const float* __restrict__ src,
                       unsigned short* __restrict__ hi,
                       unsigned short* __restrict__ lo, int n4) {
    int i = blockIdx.x * blockDim.x + threadIdx.x;
    if (i < n4) {
        float4 f = ((const float4*)src)[i];
        float v[4] = {f.x, f.y, f.z, f.w};
        v4us h, l;
#pragma unroll
        for (int j = 0; j < 4; ++j) {
            unsigned short t = f32_to_bf16(v[j]);
            h[j] = t;
            l[j] = f32_to_bf16(v[j] - bf16_to_f32(t));
        }
        ((v4us*)hi)[i] = h;
        ((v4us*)lo)[i] = l;
    }
}

// Fast grouped-GEMM: preprocessed bf16 hi/lo planes, pure VMEM + WMMA hot loop.
// grid = (COUNT experts, OUT/256 col tiles), block = 256 (8 waves).
// Each wave computes a 32x32 C tile (2x2 reps, 4 accumulators), bf16x3 accuracy.
// Per K-step: 16 global_load_b128, 12 v_wmma_f32_16x16x32_bf16.
__global__ __launch_bounds__(256)
void moe_gemm_bf16_kernel(const unsigned short* __restrict__ xhi,
                          const unsigned short* __restrict__ xlo,
                          const unsigned short* __restrict__ whi,
                          const unsigned short* __restrict__ wlo,
                          const int*   __restrict__ perm,
                          const int*   __restrict__ counts,
                          const float* __restrict__ bias,
                          float*       __restrict__ out) {
    const int e      = blockIdx.x;
    const int n_e    = counts[e];
    const int wave   = threadIdx.x >> 5;
    const int lane   = threadIdx.x & 31;
    const int laneLo = lane & 15;     // N for B/C/D; M-within-16 for A loads
    const int laneHi = lane >> 4;     // K-octet select per CDNA5 16-bit layout
    const int kbase  = laneHi * 8;
    const int col0   = blockIdx.y * 256 + wave * 32 + laneLo;

    const unsigned short* bh_p[2];
    const unsigned short* bl_p[2];
    float bcol[2];
#pragma unroll
    for (int c = 0; c < 2; ++c) {
        size_t woff = (size_t)(e * OUT_FEATURES + col0 + c * 16) * IN_FEATURES;
        bh_p[c] = whi + woff;
        bl_p[c] = wlo + woff;
        bcol[c] = bias[e * OUT_FEATURES + col0 + c * 16];
    }
    const int* pbase = perm + e * BATCH;

    __shared__ int s_rows[32];

    for (int chunk = 0; chunk * 32 < n_e; ++chunk) {
        __syncthreads();
        if (threadIdx.x < 32) {
            int m = chunk * 32 + (int)threadIdx.x;
            s_rows[threadIdx.x] = (m < n_e) ? pbase[m] : 0;
        }
        __syncthreads();

        const unsigned short* ah_p[2];
        const unsigned short* al_p[2];
#pragma unroll
        for (int r = 0; r < 2; ++r) {
            size_t row = (size_t)s_rows[r * 16 + laneLo];
            ah_p[r] = xhi + row * IN_FEATURES;
            al_p[r] = xlo + row * IN_FEATURES;
        }

        v8f acc[2][2];
#pragma unroll
        for (int r = 0; r < 2; ++r)
#pragma unroll
            for (int c = 0; c < 2; ++c) acc[r][c] = (v8f){};

        for (int k0 = 0; k0 < IN_FEATURES; k0 += 32) {
            const int off0 = k0 + kbase;       // K octets 0..7
            const int off1 = off0 + 16;        // K octets 16..23

            v16bf Bh[2], Bl[2];
#pragma unroll
            for (int c = 0; c < 2; ++c) {
                Bh[c] = mk_frag(*(const v8us*)(bh_p[c] + off0),
                                *(const v8us*)(bh_p[c] + off1));
                Bl[c] = mk_frag(*(const v8us*)(bl_p[c] + off0),
                                *(const v8us*)(bl_p[c] + off1));
            }
#pragma unroll
            for (int r = 0; r < 2; ++r) {
                v16bf Ah = mk_frag(*(const v8us*)(ah_p[r] + off0),
                                   *(const v8us*)(ah_p[r] + off1));
                v16bf Al = mk_frag(*(const v8us*)(al_p[r] + off0),
                                   *(const v8us*)(al_p[r] + off1));
#pragma unroll
                for (int c = 0; c < 2; ++c) {
                    acc[r][c] = __builtin_amdgcn_wmma_f32_16x16x32_bf16(
                        false, Ah, false, Bh[c], (short)0, acc[r][c], false, false);
                    acc[r][c] = __builtin_amdgcn_wmma_f32_16x16x32_bf16(
                        false, Ah, false, Bl[c], (short)0, acc[r][c], false, false);
                    acc[r][c] = __builtin_amdgcn_wmma_f32_16x16x32_bf16(
                        false, Al, false, Bh[c], (short)0, acc[r][c], false, false);
                }
            }
        }

#pragma unroll
        for (int r = 0; r < 2; ++r) {
#pragma unroll
            for (int j = 0; j < 8; ++j) {
                int m = r * 16 + j + 8 * laneHi;
                if (chunk * 32 + m < n_e) {
                    size_t obase = (size_t)s_rows[m] * OUT_FEATURES;
#pragma unroll
                    for (int c = 0; c < 2; ++c) {
                        out[obase + col0 + c * 16] = acc[r][c][j] + bcol[c];
                    }
                }
            }
        }
    }
}

// Fallback (small ws): in-loop conversion, 16x16 per wave.
__global__ __launch_bounds__(256)
void moe_gemm_f32_kernel(const float* __restrict__ x,
                         const int*   __restrict__ perm,
                         const int*   __restrict__ counts,
                         const float* __restrict__ W,
                         const float* __restrict__ bias,
                         float*       __restrict__ out) {
    const int e      = blockIdx.x;
    const int n_e    = counts[e];
    const int wave   = threadIdx.x >> 5;
    const int lane   = threadIdx.x & 31;
    const int laneLo = lane & 15;
    const int laneHi = lane >> 4;
    const int kbase  = laneHi * 8;
    const int col    = blockIdx.y * 128 + wave * 16 + laneLo;

    const float* Wrow  = W + (size_t)(e * OUT_FEATURES + col) * IN_FEATURES;
    const float  bcol  = bias[e * OUT_FEATURES + col];
    const int*   pbase = perm + e * BATCH;

    __shared__ int s_rows[16];

    for (int chunk = 0; chunk * 16 < n_e; ++chunk) {
        __syncthreads();
        if (threadIdx.x < 16) {
            int m = chunk * 16 + (int)threadIdx.x;
            s_rows[threadIdx.x] = (m < n_e) ? pbase[m] : 0;
        }
        __syncthreads();
        const float* xrow = x + (size_t)s_rows[laneLo] * IN_FEATURES;

        v8f acc = {};
        for (int k0 = 0; k0 < IN_FEATURES; k0 += 32) {
            float4 a0 = *(const float4*)(xrow + k0 + kbase);
            float4 a1 = *(const float4*)(xrow + k0 + kbase + 4);
            float4 a2 = *(const float4*)(xrow + k0 + kbase + 16);
            float4 a3 = *(const float4*)(xrow + k0 + kbase + 20);
            float4 b0 = *(const float4*)(Wrow + k0 + kbase);
            float4 b1 = *(const float4*)(Wrow + k0 + kbase + 4);
            float4 b2 = *(const float4*)(Wrow + k0 + kbase + 16);
            float4 b3 = *(const float4*)(Wrow + k0 + kbase + 20);
            float af[16] = {a0.x, a0.y, a0.z, a0.w, a1.x, a1.y, a1.z, a1.w,
                            a2.x, a2.y, a2.z, a2.w, a3.x, a3.y, a3.z, a3.w};
            float bf[16] = {b0.x, b0.y, b0.z, b0.w, b1.x, b1.y, b1.z, b1.w,
                            b2.x, b2.y, b2.z, b2.w, b3.x, b3.y, b3.z, b3.w};
            v16us ahi, alo, bhi, blo;
#pragma unroll
            for (int j = 0; j < 16; ++j) {
                unsigned short h = f32_to_bf16(af[j]);
                ahi[j] = h;
                alo[j] = f32_to_bf16(af[j] - bf16_to_f32(h));
                unsigned short g = f32_to_bf16(bf[j]);
                bhi[j] = g;
                blo[j] = f32_to_bf16(bf[j] - bf16_to_f32(g));
            }
            v16bf Ah = __builtin_bit_cast(v16bf, ahi);
            v16bf Al = __builtin_bit_cast(v16bf, alo);
            v16bf Bh = __builtin_bit_cast(v16bf, bhi);
            v16bf Bl = __builtin_bit_cast(v16bf, blo);
            acc = __builtin_amdgcn_wmma_f32_16x16x32_bf16(false, Ah, false, Bh,
                                                          (short)0, acc, false, false);
            acc = __builtin_amdgcn_wmma_f32_16x16x32_bf16(false, Ah, false, Bl,
                                                          (short)0, acc, false, false);
            acc = __builtin_amdgcn_wmma_f32_16x16x32_bf16(false, Al, false, Bh,
                                                          (short)0, acc, false, false);
        }
#pragma unroll
        for (int r = 0; r < 8; ++r) {
            int m = r + 8 * laneHi;
            if (chunk * 16 + m < n_e) {
                out[(size_t)s_rows[m] * OUT_FEATURES + col] = acc[r] + bcol;
            }
        }
    }
}

extern "C" void kernel_launch(void* const* d_in, const int* in_sizes, int n_in,
                              void* d_out, int out_size, void* d_ws, size_t ws_size,
                              hipStream_t stream) {
    (void)in_sizes; (void)n_in; (void)out_size;
    const float* x    = (const float*)d_in[0];
    const int*   idx  = (const int*)d_in[1];
    const float* W    = (const float*)d_in[2];
    const float* bias = (const float*)d_in[3];
    float*       out  = (float*)d_out;

    const size_t xElems = (size_t)BATCH * IN_FEATURES;                 // 8M
    const size_t wElems = (size_t)OUT_FEATURES * COUNT * IN_FEATURES;  // 16M

    char* ws = (char*)d_ws;
    int* counts = (int*)ws;                       // 64 B
    int* perm   = (int*)(ws + 256);               // 512 KB
    size_t off  = 256 + (size_t)COUNT * BATCH * sizeof(int);
    unsigned short* xhi = (unsigned short*)(ws + off);  off += xElems * 2;
    unsigned short* xlo = (unsigned short*)(ws + off);  off += xElems * 2;
    unsigned short* whi = (unsigned short*)(ws + off);  off += wElems * 2;
    unsigned short* wlo = (unsigned short*)(ws + off);  off += wElems * 2;
    const size_t need = off;

    zero_counts_kernel<<<1, 32, 0, stream>>>(counts);
    bin_rows_kernel<<<BATCH / 256, 256, 0, stream>>>(idx, counts, perm);

    if (ws_size >= need) {
        split_bf16_kernel<<<(int)(xElems / 4 / 256), 256, 0, stream>>>(x, xhi, xlo,
                                                                       (int)(xElems / 4));
        split_bf16_kernel<<<(int)(wElems / 4 / 256), 256, 0, stream>>>(W, whi, wlo,
                                                                       (int)(wElems / 4));
        dim3 grid(COUNT, OUT_FEATURES / 256);
        moe_gemm_bf16_kernel<<<grid, 256, 0, stream>>>(xhi, xlo, whi, wlo,
                                                       perm, counts, bias, out);
    } else {
        dim3 grid(COUNT, OUT_FEATURES / 128);
        moe_gemm_f32_kernel<<<grid, 256, 0, stream>>>(x, perm, counts, W, bias, out);
    }
}
